// BaselineKNNModule_19224273617288
// MI455X (gfx1250) — matrix operand order repaired and checked by reading, not compiled, not run on hardware
//
#include <hip/hip_runtime.h>
#include <hip/hip_bf16.h>
#include <float.h>

// ---------------------------------------------------------------------------
// KNN retrieval for MI455X (gfx1250):
//   scores = Q[256,128] @ corpus.T[128,N], top-100 per query, gather rows.
//
// Roofline: corpus = 512 MB fp32 must stream once -> 22 us @ 23.3 TB/s.
// bf16 WMMA (V_WMMA_F32_16X16X32_BF16, f32 accumulate) keeps the kernel
// memory-bound; fp32 WMMA (16x16x4) would be ~5x compute-bound.
// K1: per-chunk exact top-100 (disjoint chunks => global top-100 is a subset
//     of the candidate union). K2: merge 128*100 candidates/query + gather.
// ---------------------------------------------------------------------------

typedef __bf16 bf16_t;
typedef bf16_t v16bf __attribute__((ext_vector_type(16)));
typedef float  v8f   __attribute__((ext_vector_type(8)));

#define BATCH 256
#define DIM   128
#define TOPK  100
#define TN    16          // corpus rows per tile
#define SCR_STRIDE 17     // padded score-staging stride (bank-conflict free)

__global__ __launch_bounds__(256)
void knn_score_select(const float* __restrict__ Q,
                      const float* __restrict__ C,
                      float* __restrict__ candS,
                      int*   __restrict__ candI,
                      int N, int nblk, int tilesPerBlk, int totTiles)
{
    __shared__ __attribute__((aligned(32))) bf16_t s_tile[TN * DIM]; // 4 KB
    __shared__ float s_scores[BATCH * SCR_STRIDE];                   // 17.4 KB

    const int tid  = threadIdx.x;
    const int lane = tid & 31;
    const int wave = tid >> 5;
    const int half = lane >> 4;       // lane half selects K sub-range
    const int row  = lane & 15;       // A: M row / B: N column / C: N column
    const int qbase = wave * 32;      // this wave owns queries [qbase, qbase+32)

    // ---- A operands: two 16x32 bf16 query tiles per K-block, kept in VGPRs.
    // ISA 16-bit A layout: lanes 0-15 hold K={0..7,16..23}, lanes 16-31 hold
    // K={8..15,24..31} of row M = lane&15 (elements 2i,2i+1 pack VGPR i).
    v16bf A0[4], A1[4];
#pragma unroll
    for (int kb = 0; kb < 4; ++kb) {
        const int k0 = kb * 32 + half * 8;
        const int k1 = kb * 32 + 16 + half * 8;
        const float* r0 = Q + (size_t)(qbase + row) * DIM;
        const float* r1 = Q + (size_t)(qbase + 16 + row) * DIM;
        v16bf a0, a1;
#pragma unroll
        for (int i = 0; i < 8; ++i) {
            a0[i]     = (bf16_t)r0[k0 + i];
            a0[8 + i] = (bf16_t)r0[k1 + i];
            a1[i]     = (bf16_t)r1[k0 + i];
            a1[8 + i] = (bf16_t)r1[k1 + i];
        }
        A0[kb] = a0; A1[kb] = a1;
    }

    // per-thread (query == tid) candidate list for this block's chunk
    float* myS = candS + ((size_t)tid * nblk + blockIdx.x) * TOPK;
    int*   myI = candI + ((size_t)tid * nblk + blockIdx.x) * TOPK;
    int   cnt = 0, minPos = 0;
    float runMin = -FLT_MAX;

    const int tileBeg = blockIdx.x * tilesPerBlk;
    int tileEnd = tileBeg + tilesPerBlk;
    if (tileEnd > totTiles) tileEnd = totTiles;

    const int ldRow = tid >> 4;        // 0..15
    const int ldCol = (tid & 15) * 8;  // 8 floats per thread

    for (int t = tileBeg; t < tileEnd; ++t) {
        const int rowBase = t * TN;

        // ---- cooperative load + fp32->bf16 convert of a 16x128 corpus tile
        {
            const int gr = rowBase + ldRow;
            bf16_t* dst = &s_tile[ldRow * DIM + ldCol];
            if (gr < N) {
                const float* src = C + (size_t)gr * DIM + ldCol;
                __builtin_prefetch(src + TN * DIM, 0, 3);  // global_prefetch_b8
#pragma unroll
                for (int i = 0; i < 8; ++i) dst[i] = (bf16_t)src[i];
            } else {
#pragma unroll
                for (int i = 0; i < 8; ++i) dst[i] = (bf16_t)0.0f;
            }
        }
        __syncthreads();

        // ---- WMMA: scores[32q x 16n] for this wave.
        // B layout (32x16 bf16): lane holds column n = lane&15 and K values
        // kb*32 + half*8 + {0..7} and kb*32 + 16 + half*8 + {0..7} (matches A)
        // -> two aligned 16-byte LDS reads per K-block.
        v8f acc0 = {}, acc1 = {};
#pragma unroll
        for (int kb = 0; kb < 4; ++kb) {
            typedef bf16_t v8bf __attribute__((ext_vector_type(8)));
            const v8bf blo = *(const v8bf*)&s_tile[row * DIM + kb * 32 + half * 8];
            const v8bf bhi = *(const v8bf*)&s_tile[row * DIM + kb * 32 + 16 + half * 8];
            const v16bf b = __builtin_shufflevector(blo, bhi, 0, 1, 2, 3, 4, 5, 6, 7, 8, 9, 10, 11, 12, 13, 14, 15);
            acc0 = __builtin_amdgcn_wmma_f32_16x16x32_bf16(
                false, A0[kb], false, b, (short)0, acc0, false, false);
            acc1 = __builtin_amdgcn_wmma_f32_16x16x32_bf16(
                false, A1[kb], false, b, (short)0, acc1, false, false);
        }

        // ---- C/D layout: VGPR r, lane l -> M = r + 8*(l>>4), N = l&15
#pragma unroll
        for (int r = 0; r < 8; ++r) {
            s_scores[(qbase + 8 * half + r) * SCR_STRIDE + row]      = acc0[r];
            s_scores[(qbase + 16 + 8 * half + r) * SCR_STRIDE + row] = acc1[r];
        }
        __syncthreads();

        // ---- per-thread exact chunk-local top-100 (replace-min, L2-resident)
#pragma unroll
        for (int n = 0; n < TN; ++n) {
            const float s  = s_scores[tid * SCR_STRIDE + n];
            const int  idx = rowBase + n;
            if (idx >= N) continue;
            if (cnt < TOPK) {
                myS[cnt] = s; myI[cnt] = idx; ++cnt;
                if (cnt == TOPK) {
                    runMin = myS[0]; minPos = 0;
                    for (int j = 1; j < TOPK; ++j)
                        if (myS[j] < runMin) { runMin = myS[j]; minPos = j; }
                }
            } else if (s > runMin) {
                myS[minPos] = s; myI[minPos] = idx;
                runMin = myS[0]; minPos = 0;
                for (int j = 1; j < TOPK; ++j)
                    if (myS[j] < runMin) { runMin = myS[j]; minPos = j; }
            }
        }
        // No third barrier needed: next iteration's s_tile stores happen after
        // this iteration's barrier #2, and its s_scores writes happen after
        // next iteration's barrier #1 (which follows everyone's selection).
    }

    // pad unfilled slots (only when a chunk has < 100 rows)
    while (cnt < TOPK) { myS[cnt] = -FLT_MAX; myI[cnt] = 0; ++cnt; }
}

__global__ __launch_bounds__(256)
void knn_merge_gather(const float* __restrict__ C,
                      float* __restrict__ candS,
                      const int* __restrict__ candI,
                      int*   __restrict__ outIdx,
                      float* __restrict__ outVec,
                      int cpq /* candidates per query */)
{
    __shared__ float s_val[256];
    __shared__ int   s_cid[256];
    __shared__ int   s_pos[256];
    __shared__ int   s_sel[TOPK];

    const int q   = blockIdx.x;
    const int tid = threadIdx.x;
    float*     cs = candS + (size_t)q * cpq;
    const int* ci = candI + (size_t)q * cpq;

    // 100 rounds of block-wide argmax => sorted-descending indices,
    // tie-break on smaller corpus index (matches jax.lax.top_k stability).
    for (int sel = 0; sel < TOPK; ++sel) {
        float best = -FLT_MAX; int bcid = 0x7fffffff; int bpos = -1;
        for (int j = tid; j < cpq; j += 256) {
            const float s = cs[j]; const int id = ci[j];
            if (s > best || (s == best && id < bcid)) { best = s; bcid = id; bpos = j; }
        }
        s_val[tid] = best; s_cid[tid] = bcid; s_pos[tid] = bpos;
        __syncthreads();
        for (int off = 128; off > 0; off >>= 1) {
            if (tid < off) {
                const float s2 = s_val[tid + off]; const int id2 = s_cid[tid + off];
                if (s2 > s_val[tid] || (s2 == s_val[tid] && id2 < s_cid[tid])) {
                    s_val[tid] = s2; s_cid[tid] = id2; s_pos[tid] = s_pos[tid + off];
                }
            }
            __syncthreads();
        }
        if (tid == 0) {
            outIdx[(size_t)q * TOPK + sel] = s_cid[0];
            s_sel[sel] = s_cid[0];
            if (s_pos[0] >= 0) cs[s_pos[0]] = -FLT_MAX;  // remove winner
        }
        __syncthreads();
    }

    // gather the 100 selected corpus rows (100 x 128 fp32) for this query
    for (int e = tid; e < TOPK * DIM; e += 256) {
        const int r = e >> 7, c = e & (DIM - 1);
        outVec[((size_t)q * TOPK + r) * DIM + c] =
            C[(size_t)s_sel[r] * DIM + c];
    }
}

extern "C" void kernel_launch(void* const* d_in, const int* in_sizes, int n_in,
                              void* d_out, int out_size, void* d_ws, size_t ws_size,
                              hipStream_t stream)
{
    (void)n_in; (void)out_size;
    const float* Q = (const float*)d_in[0];       // [256,128] fp32
    const float* C = (const float*)d_in[1];       // [N,128]  fp32
    // d_in[2] = num_items (100 per reference setup; TOPK fixed at 100)

    const int N        = in_sizes[1] / DIM;
    const int totTiles = (N + TN - 1) / TN;

    // workspace: candS[BATCH][nblk][TOPK] f32 + candI same in i32
    int nblk = 128;
    {
        const size_t perBlk = (size_t)BATCH * TOPK * 8;  // bytes per block of chunks
        size_t maxBlk = ws_size / perBlk;
        if (maxBlk < 1) maxBlk = 1;
        if ((size_t)nblk > maxBlk) nblk = (int)maxBlk;
    }
    const int tilesPerBlk = (totTiles + nblk - 1) / nblk;

    float* candS = (float*)d_ws;
    int*   candI = (int*)(candS + (size_t)BATCH * nblk * TOPK);

    int*   outIdx = (int*)d_out;                       // [256,100] int32
    float* outVec = (float*)d_out + (size_t)BATCH * TOPK;  // [256,100,128] f32

    knn_score_select<<<nblk, 256, 0, stream>>>(Q, C, candS, candI,
                                               N, nblk, tilesPerBlk, totTiles);
    knn_merge_gather<<<BATCH, 256, 0, stream>>>(C, candS, candI,
                                                outIdx, outVec, nblk * TOPK);
}
